// SparseCrossAttention_14491219657194
// MI455X (gfx1250) — matrix operand order repaired and checked by reading, not compiled, not run on hardware
//
#include <hip/hip_runtime.h>
#include <hip/hip_bf16.h>
#include <cstdint>

// Problem constants (reference: B=32, N=S=2048, D=1024, H=16, dh=64, TOPK=10)
#define BB 32
#define NN 2048
#define SS 2048
#define DD 1024
#define HH 16
#define DH 64
#define TK 10
#define ST 32               // scaffold rows per tile (2 WMMA row-groups)

typedef __attribute__((ext_vector_type(16))) __bf16 v16bf;
typedef __attribute__((ext_vector_type(8)))  float  v8f;

// ---------------------------------------------------------------- scores ----
__global__ __launch_bounds__(256) void sca_scores(
    const float* __restrict__ base, const float* __restrict__ score_w,
    const float* __restrict__ score_b, float* __restrict__ scores) {
  const int row  = blockIdx.x * 8 + (threadIdx.x >> 5);   // 8 waves/block
  const int lane = threadIdx.x & 31;
  const float4* r = (const float4*)(base + (size_t)row * DD);
  const float4* w = (const float4*)score_w;
  float s = 0.f;
  for (int i = lane; i < DD / 4; i += 32) {
    float4 a = r[i], b = w[i];
    s += a.x * b.x + a.y * b.y + a.z * b.z + a.w * b.w;
  }
#pragma unroll
  for (int m = 16; m >= 1; m >>= 1) s += __shfl_xor(s, m, 32);
  if (lane == 0) scores[row] = s + score_b[0];
}

// ----------------------------------------------------------------- top-k ----
__global__ __launch_bounds__(256) void sca_topk(
    const float* __restrict__ scores, int* __restrict__ topk) {
  const int b = blockIdx.x, tid = threadIdx.x;
  __shared__ float sv[NN];
  __shared__ float rv[256];
  __shared__ int   ri[256];
  for (int i = tid; i < NN; i += 256) sv[i] = scores[b * NN + i];
  __syncthreads();
  for (int k = 0; k < TK; ++k) {
    float bv = -3.4e38f; int bi = 0x7fffffff;
    for (int i = tid; i < NN; i += 256) {
      float v = sv[i];
      if (v > bv || (v == bv && i < bi)) { bv = v; bi = i; }
    }
    rv[tid] = bv; ri[tid] = bi;
    __syncthreads();
    for (int off = 128; off > 0; off >>= 1) {
      if (tid < off) {
        float v2 = rv[tid + off]; int i2 = ri[tid + off];
        if (v2 > rv[tid] || (v2 == rv[tid] && i2 < ri[tid])) { rv[tid] = v2; ri[tid] = i2; }
      }
      __syncthreads();
    }
    if (tid == 0) { topk[b * TK + k] = ri[0]; sv[ri[0]] = -3.4e38f; }
    __syncthreads();
  }
}

// ------------------------------------------------------------------ copy ----
__global__ __launch_bounds__(256) void sca_copy(
    const float4* __restrict__ src, float4* __restrict__ dst) {
  const unsigned total = (unsigned)(BB * NN * DD / 4);
  unsigned stride = gridDim.x * 256u;
  for (unsigned i = blockIdx.x * 256u + threadIdx.x; i < total; i += stride)
    dst[i] = src[i];
}

// ----------------------------------------------------- weight repack bf16 ---
// Wk/Wv -> bf16 in WMMA B-matrix (32x16, 16-bit) lane layout, contiguous k order (A uses the same):
// tile(mat,h,ntile,kstep): lane l = column n=l&15; 16 consecutive K from (l>>4)*16.
__global__ __launch_bounds__(256) void sca_repack(
    const float* __restrict__ ipw, __bf16* __restrict__ wp) {
  const unsigned id   = blockIdx.x * 256u + threadIdx.x;   // < 2,097,152
  const unsigned e    = id & 15u;
  const unsigned lane = (id >> 4) & 31u;
  const unsigned ks   = (id >> 9) & 31u;
  const unsigned t    = (id >> 14) & 3u;
  const unsigned h    = (id >> 16) & 15u;
  const unsigned mat  = (id >> 20) & 1u;                   // 0=K, 1=V
  const unsigned n    = lane & 15u, kh = lane >> 4;
  const unsigned row  = (1u + mat) * DD + h * DH + t * 16u + n;
  const unsigned col  = ks * 32u + kh * 16u + e;
  wp[id] = (__bf16)ipw[(size_t)row * DD + col];
}

// ---------------------------------------------------------- Q projection ----
__global__ __launch_bounds__(256) void sca_qproj(
    const float* __restrict__ base, const float* __restrict__ ipw,
    const float* __restrict__ ipb, const int* __restrict__ topk,
    float* __restrict__ qbuf) {
  __shared__ float4 xs[DD / 4];
  const int b = blockIdx.x / TK, qk = blockIdx.x - b * TK;
  const int tid = threadIdx.x;
  const int idx = topk[b * TK + qk];
  const float4* x = (const float4*)(base + (size_t)(b * NN + idx) * DD);
  xs[tid] = x[tid];
  __syncthreads();
#pragma unroll
  for (int rep = 0; rep < 4; ++rep) {
    const int d = tid + rep * 256;
    const float4* w = (const float4*)(ipw + (size_t)d * DD);   // Wq rows 0..D-1
    float acc = ipb[d];
    for (int k = 0; k < DD / 4; ++k) {
      float4 a = xs[k], wv = w[k];
      acc += a.x * wv.x + a.y * wv.y + a.z * wv.z + a.w * wv.w;
    }
    qbuf[(b * TK + qk) * DD + d] = acc;
  }
}

// ---------------------------------------------- fused K/V proj + attention --
// One workgroup per (b, h); 4 waves.  32-row scaffold tiles: each wave owns one
// 16-column slice (ntile) and computes K/V for BOTH 16-row groups, so every
// B-matrix (weight) load feeds 2 WMMAs per matrix -> halved L2 weight traffic.
__global__ __launch_bounds__(128) void sca_fused(
    const float* __restrict__ scaffold, const float* __restrict__ ipb,
    const float* __restrict__ qbuf, const __bf16* __restrict__ wpack,
    float* __restrict__ obuf) {
  const int b = blockIdx.x >> 4, h = blockIdx.x & 15;
  const int tid = threadIdx.x, lane = tid & 31, wave = tid >> 5;
  const int m  = lane & 15;      // A-row / C-column index
  const int kh = lane >> 4;      // K-half / C row-group

  __shared__ __bf16 Asc[ST][1032];          // 66 KB scaffold tile (bf16)
  __shared__ float  Qs[TK][DH];
  __shared__ float  Ks[ST][DH + 4];
  __shared__ float  Vs[ST][DH + 4];
  __shared__ float  Pl[TK][ST];
  __shared__ float  Mrow[TK], Lrow[TK], Srow[TK];

  for (int i = tid; i < TK * DH; i += 128) {
    int q = i >> 6, d = i & 63;
    Qs[q][d] = qbuf[(b * TK + q) * DD + h * DH + d];
  }
  if (tid < TK) { Mrow[tid] = -3.4e38f; Lrow[tid] = 0.f; }
  float oacc[5];
#pragma unroll
  for (int i = 0; i < 5; ++i) oacc[i] = 0.f;

  const int ccol = h * DH + wave * 16 + m;
  const float biask = ipb[DD + ccol];
  const float biasv = ipb[2 * DD + ccol];
  const __bf16* wk_base = wpack + ((size_t)h << 16) + ((size_t)wave << 14);
  const __bf16* wv_base = wk_base + (1u << 20);
  __syncthreads();

  for (int s0 = 0; s0 < SS; s0 += ST) {
    // ---- stage scaffold tile [ST x 1024] -> LDS bf16 (coalesced) ----
    const float4* src = (const float4*)(scaffold + (size_t)(b * SS + s0) * DD);
    if (s0 + ST < SS) {
      const float* nxt = scaffold + (size_t)(b * SS + s0 + ST) * DD;
      for (int p = tid; p < ST * 32; p += 128) __builtin_prefetch(nxt + p * 32, 0, 1);
    }
    for (int fi = tid; fi < ST * 256; fi += 128) {
      int row = fi >> 8, c4 = fi & 255;
      float4 v = src[row * 256 + c4];
      __bf16* dp = &Asc[row][c4 * 4];
      dp[0] = (__bf16)v.x; dp[1] = (__bf16)v.y; dp[2] = (__bf16)v.z; dp[3] = (__bf16)v.w;
    }
    __syncthreads();

    // ---- K & V projection: 2 row-groups share each weight tile ----
    v8f ack0 = {}, ack1 = {}, acv0 = {}, acv1 = {};
#pragma unroll 2
    for (int ks = 0; ks < 32; ++ks) {
      union { v16bf v; unsigned u[8]; } A0, A1;
      const unsigned* rp0 = (const unsigned*)&Asc[m][ks * 32];
      const unsigned* rp1 = (const unsigned*)&Asc[16 + m][ks * 32];
#pragma unroll
      for (int i = 0; i < 8; ++i) {                     // A: contiguous k = kh*16 + e, as in B pack
        int kb = kh * 16 + 2 * i;
        A0.u[i] = rp0[kb >> 1];
        A1.u[i] = rp1[kb >> 1];
      }
      v16bf bk = *(const v16bf*)(wk_base + ks * 512 + lane * 16);
      v16bf bv = *(const v16bf*)(wv_base + ks * 512 + lane * 16);
      ack0 = __builtin_amdgcn_wmma_f32_16x16x32_bf16(false, A0.v, false, bk, (short)0, ack0, false, false);
      acv0 = __builtin_amdgcn_wmma_f32_16x16x32_bf16(false, A0.v, false, bv, (short)0, acv0, false, false);
      ack1 = __builtin_amdgcn_wmma_f32_16x16x32_bf16(false, A1.v, false, bk, (short)0, ack1, false, false);
      acv1 = __builtin_amdgcn_wmma_f32_16x16x32_bf16(false, A1.v, false, bv, (short)0, acv1, false, false);
    }
    {
      const int col = wave * 16 + m;                    // C/D layout: n = lane&15
#pragma unroll
      for (int r = 0; r < 8; ++r) {
        int s = r + 8 * kh;                             // row = r + 8*(lane>>4)
        Ks[s][col]      = ack0[r] + biask;
        Vs[s][col]      = acv0[r] + biasv;
        Ks[16 + s][col] = ack1[r] + biask;
        Vs[16 + s][col] = acv1[r] + biasv;
      }
    }
    __syncthreads();

    // ---- logits (TK x ST), scale 1/sqrt(64) ----
    for (int p = tid; p < TK * ST; p += 128) {
      int q = p >> 5, s = p & 31;
      float acc = 0.f;
#pragma unroll
      for (int d = 0; d < DH; ++d) acc += Qs[q][d] * Ks[s][d];
      Pl[q][s] = acc * 0.125f;
    }
    __syncthreads();

    // ---- online softmax row stats ----
    if (tid < TK) {
      int q = tid;
      float rmax = Pl[q][0];
#pragma unroll
      for (int s = 1; s < ST; ++s) rmax = fmaxf(rmax, Pl[q][s]);
      float mo = Mrow[q], mn = fmaxf(mo, rmax);
      float sc = expf(mo - mn);
      float lsum = 0.f;
#pragma unroll
      for (int s = 0; s < ST; ++s) { float pv = expf(Pl[q][s] - mn); Pl[q][s] = pv; lsum += pv; }
      Mrow[q] = mn; Srow[q] = sc; Lrow[q] = Lrow[q] * sc + lsum;
    }
    __syncthreads();

    // ---- O accumulation: thread owns 5 (q,d) slots ----
#pragma unroll
    for (int i = 0; i < 5; ++i) {
      int pr = tid + 128 * i, q = pr >> 6, d = pr & 63;
      float acc = oacc[i] * Srow[q];
#pragma unroll
      for (int s = 0; s < ST; ++s) acc += Pl[q][s] * Vs[s][d];
      oacc[i] = acc;
    }
    __syncthreads();
  }

#pragma unroll
  for (int i = 0; i < 5; ++i) {
    int pr = tid + 128 * i, q = pr >> 6, d = pr & 63;
    obuf[(b * TK + q) * DD + h * DH + d] = oacc[i] / Lrow[q];
  }
}

// -------------------------------------------- out projection + scatter ------
__global__ __launch_bounds__(256) void sca_outproj(
    const float* __restrict__ obuf, const float* __restrict__ outw,
    const float* __restrict__ outb, const int* __restrict__ topk,
    float* __restrict__ out) {
  __shared__ float4 xs[DD / 4];
  const int b = blockIdx.x / TK, qk = blockIdx.x - b * TK;
  const int tid = threadIdx.x;
  const int idx = topk[b * TK + qk];
  const float4* x = (const float4*)(obuf + (size_t)(b * TK + qk) * DD);
  xs[tid] = x[tid];
  __syncthreads();
  float* drow = out + (size_t)(b * NN + idx) * DD;
#pragma unroll
  for (int rep = 0; rep < 4; ++rep) {
    const int d = tid + rep * 256;
    const float4* w = (const float4*)(outw + (size_t)d * DD);
    float acc = outb[d];
    for (int k = 0; k < DD / 4; ++k) {
      float4 a = xs[k], wv = w[k];
      acc += a.x * wv.x + a.y * wv.y + a.z * wv.z + a.w * wv.w;
    }
    drow[d] = acc;
  }
}

// ----------------------------------------------------------------- launch ---
extern "C" void kernel_launch(void* const* d_in, const int* in_sizes, int n_in,
                              void* d_out, int out_size, void* d_ws, size_t ws_size,
                              hipStream_t stream) {
  (void)in_sizes; (void)n_in; (void)out_size; (void)ws_size;
  const float* base = (const float*)d_in[0];
  const float* scaf = (const float*)d_in[1];
  const float* ipw  = (const float*)d_in[2];
  const float* ipb  = (const float*)d_in[3];
  const float* outw = (const float*)d_in[4];
  const float* outb = (const float*)d_in[5];
  const float* scw  = (const float*)d_in[6];
  const float* scb  = (const float*)d_in[7];
  float* out = (float*)d_out;

  char* ws = (char*)d_ws;
  float*  scores = (float*)(ws + 0);             //  262144 B
  int*    topk   = (int*)(ws + 262144);          //    1280 B
  float*  qbuf   = (float*)(ws + 266240);        // 1310720 B
  float*  obuf   = (float*)(ws + 1576960);       // 1310720 B
  __bf16* wpack  = (__bf16*)(ws + 2887680);      // 4194304 B   (total ~6.8 MB)

  sca_scores <<<dim3(BB * NN / 8), dim3(256), 0, stream>>>(base, scw, scb, scores);
  sca_repack <<<dim3(8192),        dim3(256), 0, stream>>>(ipw, wpack);
  sca_copy   <<<dim3(4096),        dim3(256), 0, stream>>>((const float4*)base, (float4*)out);
  sca_topk   <<<dim3(BB),          dim3(256), 0, stream>>>(scores, topk);
  sca_qproj  <<<dim3(BB * TK),     dim3(256), 0, stream>>>(base, ipw, ipb, topk, qbuf);
  sca_fused  <<<dim3(BB * HH),     dim3(128), 0, stream>>>(scaf, ipb, qbuf, wpack, obuf);
  sca_outproj<<<dim3(BB * TK),     dim3(256), 0, stream>>>(obuf, outw, outb, topk, out);
}
